// SpatialLocalSum_42442866819598
// MI455X (gfx1250) — matrix operand order repaired and checked by reading, not compiled, not run on hardware
//
#include <hip/hip_runtime.h>
#include <hip/hip_bf16.h>

typedef __attribute__((ext_vector_type(16))) _Float16 v16h;
typedef __attribute__((ext_vector_type(8)))  _Float16 v8h;
typedef __attribute__((ext_vector_type(8)))  float    v8f;

// Problem constants: B=32, V=32, H=32, CIN=256, S=256
#define VH       1024          // V*H
#define KDIM     256
#define SDIM     256
#define BDIM     32

#define WT_STRIDE 260          // f16 elems per Wt row (520 B, padded vs 512 to dodge bank conflicts)
#define A_STRIDE  264          // f16 elems per A row  (528 B, padded)

// LDS layout (bytes)
#define OFF_WT    0
#define SZ_WT     (SDIM * WT_STRIDE * 2)      // 133120: Wt[s][k] f16 (transposed W)
#define OFF_A     (OFF_WT + SZ_WT)
#define SZ_A      (BDIM * A_STRIDE * 2)       // 16896:  A[b][k] f16
#define OFF_XMAX  (OFF_A + SZ_A)
#define SZ_XMAX   (BDIM * 4)                  // 128
#define OFF_WTERM (OFF_XMAX + SZ_XMAX)
#define SZ_WTERM  (SDIM * 4)                  // 1024
#define OFF_RCP   (OFF_WTERM + SZ_WTERM)
#define SZ_RCP    (SDIM * 4)                  // 1024
#define SMEM_BYTES (OFF_RCP + SZ_RCP)         // 152192 (< 160 KB -> 2 blocks/WGP)
// pass-1 reduction scratch overlays Wt region (used before Wt is written)
#define OFF_PMAX  0
#define OFF_PSUM  4096

union FragU { v16h v; v8h h[2]; unsigned long long q[4]; };

// A fragment (16x32 f16, M x K), lane K order used here (same as load_b_frag):
// lanes 0-15 hold K [k0..k0+15] contiguous with k0 = 32*kb
// lanes 16-31 the same but shifted by +16 (handled by caller via k0)
__device__ __forceinline__ v16h load_a_frag(const _Float16* Afr, int row, int k0) {
    FragU u;
    u.h[0] = *(const v8h*)(Afr + row * A_STRIDE + k0);        // 16B aligned
    u.h[1] = *(const v8h*)(Afr + row * A_STRIDE + k0 + 8);
    return u.v;
}

// B fragment (32x16 f16, K x N): lane holds 16 consecutive K for its column ->
// contiguous 32 bytes in the transposed Wt[s][k] layout (8B aligned -> 4x u64).
__device__ __forceinline__ v16h load_b_frag(const _Float16* Wt, int col, int k0) {
    const unsigned long long* p =
        (const unsigned long long*)(Wt + col * WT_STRIDE + k0);
    FragU u;
    u.q[0] = p[0]; u.q[1] = p[1]; u.q[2] = p[2]; u.q[3] = p[3];
    return u.v;
}

__global__ __launch_bounds__(256) void logmatmul_vhs_kernel(
    const float* __restrict__ x,      // (B,V,H,CIN)
    const float* __restrict__ acc,    // (V,H,CIN,S)
    float* __restrict__ out)          // (B,V,H,S)
{
    extern __shared__ char smem[];
    _Float16* Wt   = (_Float16*)(smem + OFF_WT);
    _Float16* Afr  = (_Float16*)(smem + OFF_A);
    float*    xmx  = (float*)(smem + OFF_XMAX);
    float*    wtm  = (float*)(smem + OFF_WTERM);
    float*    rcp  = (float*)(smem + OFF_RCP);
    float4*   pmax = (float4*)(smem + OFF_PMAX);
    float4*   psum = (float4*)(smem + OFF_PSUM);

    const int tid  = threadIdx.x;
    const int lane = tid & 31;
    const int wave = tid >> 5;
    const int vh   = blockIdx.x;              // v*H + h

    const float4* accv = (const float4*)(acc + (size_t)vh * (KDIM * SDIM));

    // ---------- Pass 1: column (s) max & sum of acc over k ----------
    const int cg = tid & 63;                  // column group: s = 4*cg .. 4*cg+3
    const int kr = tid >> 6;                  // k quarter: rows kr*64 .. kr*64+63
    float4 cmax = make_float4(0.f, 0.f, 0.f, 0.f);   // acc > 0
    float4 csum = make_float4(0.f, 0.f, 0.f, 0.f);
    for (int j = 0; j < 64; ++j) {
        const int k = kr * 64 + j;
        if (j + 8 < 64) __builtin_prefetch(&accv[(k + 8) * 64 + cg], 0, 0);
        const float4 a = accv[k * 64 + cg];   // coalesced: 512B per wave
        cmax.x = fmaxf(cmax.x, a.x); cmax.y = fmaxf(cmax.y, a.y);
        cmax.z = fmaxf(cmax.z, a.z); cmax.w = fmaxf(cmax.w, a.w);
        csum.x += a.x; csum.y += a.y; csum.z += a.z; csum.w += a.w;
    }
    pmax[kr * 64 + cg] = cmax;
    psum[kr * 64 + cg] = csum;
    __syncthreads();

    if (tid < 64) {
        float4 m = pmax[tid], s = psum[tid];
        #pragma unroll
        for (int r = 1; r < 4; ++r) {
            const float4 m2 = pmax[r * 64 + tid], s2 = psum[r * 64 + tid];
            m.x = fmaxf(m.x, m2.x); m.y = fmaxf(m.y, m2.y);
            m.z = fmaxf(m.z, m2.z); m.w = fmaxf(m.w, m2.w);
            s.x += s2.x; s.y += s2.y; s.z += s2.z; s.w += s2.w;
        }
        float4 rc = make_float4(1.f / m.x, 1.f / m.y, 1.f / m.z, 1.f / m.w);
        ((float4*)rcp)[tid] = rc;
        float4 wv = make_float4(__logf(m.x) - __logf(s.x),
                                __logf(m.y) - __logf(s.y),
                                __logf(m.z) - __logf(s.z),
                                __logf(m.w) - __logf(s.w));
        ((float4*)wtm)[tid] = wv;            // w_max[s] = log(colmax) - log(colsum)
    }

    // ---------- x pass: rowmax + A = f16(exp(x - rowmax)) ----------
    // (independent of pass-1 scratch; overlaps the tid<64 combine)
    #pragma unroll
    for (int r = 0; r < 4; ++r) {
        const int b = wave * 4 + r;
        const float* xrow = x + ((size_t)b * VH + vh) * KDIM;
        float vals[8];
        float m = -3.402823466e+38f;
        #pragma unroll
        for (int j = 0; j < 8; ++j) {
            vals[j] = xrow[lane + 32 * j];
            m = fmaxf(m, vals[j]);
        }
        #pragma unroll
        for (int off = 16; off > 0; off >>= 1)
            m = fmaxf(m, __shfl_xor(m, off, 32));   // wave32 reduction
        if (lane == 0) xmx[b] = m;
        #pragma unroll
        for (int j = 0; j < 8; ++j)
            Afr[b * A_STRIDE + lane + 32 * j] = (_Float16)__expf(vals[j] - m);
    }
    __syncthreads();

    // ---------- Pass 2: Wt[s][k] = f16(acc[k][s] / colmax[s]), L2-resident re-read ----------
    {
        const float4 rc = ((const float4*)rcp)[cg];
        for (int j = 0; j < 64; ++j) {
            const int k = kr * 64 + j;
            const float4 a = accv[k * 64 + cg];
            Wt[(4 * cg + 0) * WT_STRIDE + k] = (_Float16)(a.x * rc.x);
            Wt[(4 * cg + 1) * WT_STRIDE + k] = (_Float16)(a.y * rc.y);
            Wt[(4 * cg + 2) * WT_STRIDE + k] = (_Float16)(a.z * rc.z);
            Wt[(4 * cg + 3) * WT_STRIDE + k] = (_Float16)(a.w * rc.w);
        }
    }
    __syncthreads();

    // ---------- WMMA phase: 32x256 output = A(32x256) x W(256x256), per wave: 32-col strip ----------
    const int colBase = wave * 32;
    const int nIn = lane & 15;
    const int hi  = (lane >> 4) & 1;          // lanes 16-31 hold shifted K chunks
    v8f c00 = {}, c01 = {}, c10 = {}, c11 = {};

    for (int kb = 0; kb < 8; ++kb) {
        const int k0  = kb * 32;
        const int ka  = k0 + hi * 16;         // A-frag K start for this lane half
        const int kbb = k0 + hi * 16;         // B-frag K start for this lane half
        const v16h a0 = load_a_frag(Afr, nIn,      ka);
        const v16h a1 = load_a_frag(Afr, nIn + 16, ka);
        const v16h b0 = load_b_frag(Wt, colBase + nIn,      kbb);
        const v16h b1 = load_b_frag(Wt, colBase + 16 + nIn, kbb);
        c00 = __builtin_amdgcn_wmma_f32_16x16x32_f16(false, a0, false, b0, (short)0, c00, false, false);
        c01 = __builtin_amdgcn_wmma_f32_16x16x32_f16(false, a0, false, b1, (short)0, c01, false, false);
        c10 = __builtin_amdgcn_wmma_f32_16x16x32_f16(false, a1, false, b0, (short)0, c10, false, false);
        c11 = __builtin_amdgcn_wmma_f32_16x16x32_f16(false, a1, false, b1, (short)0, c11, false, false);
    }

    // ---------- Epilogue: out[b,vh,s] = log(dot) + xmax[b] + wterm[s] ----------
    // C/D layout: lanes 0-15 -> N=lane, M=vg; lanes 16-31 -> N=lane-16, M=8+vg
    {
        v8f ctile[4] = { c00, c01, c10, c11 };
        #pragma unroll
        for (int t = 0; t < 4; ++t) {
            const int mtile = t >> 1, ntile = t & 1;
            const int s  = colBase + ntile * 16 + nIn;
            const float wterm = wtm[s];
            const int bBase = mtile * 16 + hi * 8;
            #pragma unroll
            for (int vg = 0; vg < 8; ++vg) {
                const int b = bBase + vg;
                const float val = __logf(ctile[t][vg]) + xmx[b] + wterm;
                out[((size_t)b * VH + vh) * SDIM + s] = val;
            }
        }
    }
}

extern "C" void kernel_launch(void* const* d_in, const int* in_sizes, int n_in,
                              void* d_out, int out_size, void* d_ws, size_t ws_size,
                              hipStream_t stream) {
    (void)in_sizes; (void)n_in; (void)out_size; (void)d_ws; (void)ws_size;
    const float* x   = (const float*)d_in[0];   // (32,32,32,256) f32
    const float* acc = (const float*)d_in[1];   // (32,32,256,256) f32
    float* out = (float*)d_out;                 // (32,32,32,256) f32

    // >64KB dynamic LDS needs the opt-in attribute (idempotent; capture-safe: not a stream op)
    hipFuncSetAttribute((const void*)logmatmul_vhs_kernel,
                        hipFuncAttributeMaxDynamicSharedMemorySize, SMEM_BYTES);

    logmatmul_vhs_kernel<<<VH, 256, SMEM_BYTES, stream>>>(x, acc, out);
}
